// Attention_46746424049941
// MI455X (gfx1250) — compile-verified, matrix operand order checked by reading, not run on hardware
//
#include <hip/hip_runtime.h>
#include <hip/hip_bf16.h>

typedef __attribute__((ext_vector_type(16))) _Float16 v16h;
typedef __attribute__((ext_vector_type(8)))  _Float16 v8h;
typedef __attribute__((ext_vector_type(8)))  float    v8f;
typedef __attribute__((ext_vector_type(4)))  unsigned int u32x4;
typedef __attribute__((ext_vector_type(8)))  int          i32x8;
typedef __attribute__((ext_vector_type(4)))  int          i32x4;

// ---------------------------------------------------------------------------
// WMMA helper (CDNA5 / gfx1250, wave32)
// ---------------------------------------------------------------------------
__device__ __forceinline__ v8f wmma_f16(v16h a, v16h b, v8f c) {
    return __builtin_amdgcn_wmma_f32_16x16x32_f16(false, a, false, b,
                                                  (short)0, c, false, false);
}

// 16x32 f16 fragment load (A-layout; B-layout when source is N-major).
// lane L: row m=L%16; e=0..7 -> K=e+8*(L/16); e=8..15 -> K=16+(e-8)+8*(L/16).
__device__ __forceinline__ v16h load_frag(const _Float16* base, int stride) {
    const int lane = threadIdx.x & 31;
    const int m    = lane & 15;
    const int hi   = (lane >> 4) << 3;
    const _Float16* p = base + m * stride + hi;
    v8h lo = *(const v8h*)(p);
    v8h hb = *(const v8h*)(p + 16);
    v16h o;
#pragma unroll
    for (int i = 0; i < 8; ++i) { o[i] = lo[i]; o[i + 8] = hb[i]; }
    return o;
}

// ---------------------------------------------------------------------------
// DPP16 row_xmask reductions over the 16-lane half-row (pure VALU, no LDS)
// ---------------------------------------------------------------------------
__device__ __forceinline__ float dpp_x(float v, const int ctrl_is /*unused*/) { return v; }
__device__ __forceinline__ float xmax16(float v) {
    v = fmaxf(v, __int_as_float(__builtin_amdgcn_update_dpp(0, __float_as_int(v), 0x161, 0xf, 0xf, true)));
    v = fmaxf(v, __int_as_float(__builtin_amdgcn_update_dpp(0, __float_as_int(v), 0x162, 0xf, 0xf, true)));
    v = fmaxf(v, __int_as_float(__builtin_amdgcn_update_dpp(0, __float_as_int(v), 0x164, 0xf, 0xf, true)));
    v = fmaxf(v, __int_as_float(__builtin_amdgcn_update_dpp(0, __float_as_int(v), 0x168, 0xf, 0xf, true)));
    return v;
}
__device__ __forceinline__ float xsum16(float v) {
    v += __int_as_float(__builtin_amdgcn_update_dpp(0, __float_as_int(v), 0x161, 0xf, 0xf, true));
    v += __int_as_float(__builtin_amdgcn_update_dpp(0, __float_as_int(v), 0x162, 0xf, 0xf, true));
    v += __int_as_float(__builtin_amdgcn_update_dpp(0, __float_as_int(v), 0x164, 0xf, 0xf, true));
    v += __int_as_float(__builtin_amdgcn_update_dpp(0, __float_as_int(v), 0x168, 0xf, 0xf, true));
    return v;
}

// ---------------------------------------------------------------------------
// Tensor Data Mover: 2-D tile load Global -> LDS with LDS-side padding.
// Descriptor per CDNA5 ISA. All values uniform.
// data_size = 2 bytes (f16). pad_interval: 1<<(pi) 8-byte units between pads;
// pad_amount: (pa+1) DWORDs inserted.
// ---------------------------------------------------------------------------
__device__ __forceinline__ void tdm_load_2d(unsigned lds_off, const void* gptr,
                                            unsigned tensor_d0, unsigned tensor_d1,
                                            unsigned d0_stride,
                                            unsigned tile_d0, unsigned tile_d1,
                                            unsigned pad_interval, unsigned pad_amount) {
    const unsigned long long ga = (unsigned long long)gptr;
    u32x4 g0;
    g0[0] = 1u;                                            // count=1, user desc
    g0[1] = lds_off;                                       // lds_addr
    g0[2] = (unsigned)(ga & 0xFFFFFFFFu);                  // global_addr lo
    g0[3] = (unsigned)((ga >> 32) & 0x1FFFFFFu) | 0x80000000u; // addr hi | type=2
    i32x8 g1;
    g1[0] = (int)((1u << 16) | (1u << 20) |                // data_size=2B, pad_en
                  (pad_interval << 22) | (pad_amount << 25));
    g1[1] = (int)((tensor_d0 & 0xFFFFu) << 16);            // dim0[15:0]
    g1[2] = (int)(((tensor_d0 >> 16) & 0xFFFFu) | ((tensor_d1 & 0xFFFFu) << 16));
    g1[3] = (int)(((tensor_d1 >> 16) & 0xFFFFu) | ((tile_d0 & 0xFFFFu) << 16));
    g1[4] = (int)(tile_d1 & 0xFFFFu);                      // tile_dim1, tile_dim2=0
    g1[5] = (int)d0_stride;                                // dim0_stride[31:0]
    g1[6] = 0;
    g1[7] = 0;
    const i32x4 z = {0, 0, 0, 0};
#if defined(__clang_major__) && (__clang_major__ >= 23)
    const i32x8 z8 = {0, 0, 0, 0, 0, 0, 0, 0};
    __builtin_amdgcn_tensor_load_to_lds(g0, g1, z, z, z8, 0);
#else
    __builtin_amdgcn_tensor_load_to_lds(g0, g1, z, z, 0);
#endif
}

__device__ __forceinline__ unsigned lds_off_u32(const void* p) {
    return (unsigned)(unsigned long long)(size_t)p;        // low 32 bits = LDS offset
}

// ---------------------------------------------------------------------------
// 1) LayerNorm: x[4096,1024] f32 -> xn f16
// ---------------------------------------------------------------------------
__global__ __launch_bounds__(256)
void ln_kernel(const float* __restrict__ x, const float* __restrict__ gamma,
               const float* __restrict__ beta, _Float16* __restrict__ xn) {
    __shared__ float red[256];
    const int row = blockIdx.x;
    const int tid = threadIdx.x;
    const float* xr = x + (size_t)row * 1024;
    float v[4];
    float s = 0.f;
#pragma unroll
    for (int i = 0; i < 4; ++i) { v[i] = xr[tid + i * 256]; s += v[i]; }
    red[tid] = s; __syncthreads();
    for (int d = 128; d > 0; d >>= 1) {
        if (tid < d) red[tid] += red[tid + d];
        __syncthreads();
    }
    const float mu = red[0] * (1.f / 1024.f);
    __syncthreads();
    float ss = 0.f;
#pragma unroll
    for (int i = 0; i < 4; ++i) { float df = v[i] - mu; ss += df * df; }
    red[tid] = ss; __syncthreads();
    for (int d = 128; d > 0; d >>= 1) {
        if (tid < d) red[tid] += red[tid + d];
        __syncthreads();
    }
    const float rstd = rsqrtf(red[0] * (1.f / 1024.f) + 1e-5f);
#pragma unroll
    for (int i = 0; i < 4; ++i) {
        const int c = tid + i * 256;
        xn[(size_t)row * 1024 + c] = (_Float16)((v[i] - mu) * rstd * gamma[c] + beta[c]);
    }
}

// ---------------------------------------------------------------------------
// 2) Transpose + f32->f16: out[n*1024 + k] = in[k*Ncols + n]
// ---------------------------------------------------------------------------
__global__ __launch_bounds__(256)
void transpose_to_f16(const float* __restrict__ in, _Float16* __restrict__ out,
                      int Ncols) {
    const size_t idx = (size_t)blockIdx.x * 256 + threadIdx.x;
    const int n = (int)(idx >> 10);
    const int k = (int)(idx & 1023);
    out[idx] = (_Float16)in[(size_t)k * Ncols + n];
}

// ---------------------------------------------------------------------------
// 3/5) WMMA GEMM with TDM double-buffered staging.
//   C[M,N] = A[M,1024] * Bt[N,1024]^T, 128x128 block, 8 waves of 64x32.
//   MODE 0: QKV epilogue (Q scaled, V written transposed as Vt[b,h,d,n]).
//   MODE 1: f32 store to Out.
// ---------------------------------------------------------------------------
template <int MODE>
__global__ __launch_bounds__(256)
void gemm_wmma(const _Float16* __restrict__ A, const _Float16* __restrict__ Bt,
               _Float16* __restrict__ Qb, _Float16* __restrict__ Kb,
               _Float16* __restrict__ Vt, float* __restrict__ Out) {
    __shared__ __align__(16) _Float16 lA[2][128 * 40];   // 32-half rows + 8 pad
    __shared__ __align__(16) _Float16 lB[2][128 * 40];

    const int tid  = threadIdx.x;
    const int wave = tid >> 5;
    const int lane = tid & 31;
    const int bm   = blockIdx.y * 128;
    const int bn   = blockIdx.x * 128;
    const int wm   = (wave >> 2) * 64;
    const int wn   = (wave & 3) * 32;

    v8f acc[4][2] = {{}};

    // prologue: tile 0 in flight (row=32 halfs=64B=8x8B units -> pi=3, 16B pad -> pa=3)
    if (wave == 0) {
        tdm_load_2d(lds_off_u32(&lA[0][0]), A + (size_t)bm * 1024,
                    1024, 4096, 1024, 32, 128, 3, 3);
        tdm_load_2d(lds_off_u32(&lB[0][0]), Bt + (size_t)bn * 1024,
                    1024, 4096, 1024, 32, 128, 3, 3);
    }

    for (int it = 0; it < 32; ++it) {
        if (wave == 0) {
            if (it + 1 < 32) {
                const int k1 = (it + 1) * 32;
                tdm_load_2d(lds_off_u32(&lA[(it + 1) & 1][0]),
                            A + (size_t)bm * 1024 + k1, 1024, 4096, 1024, 32, 128, 3, 3);
                tdm_load_2d(lds_off_u32(&lB[(it + 1) & 1][0]),
                            Bt + (size_t)bn * 1024 + k1, 1024, 4096, 1024, 32, 128, 3, 3);
                __builtin_amdgcn_s_wait_tensorcnt(2);   // tile `it` complete
            } else {
                __builtin_amdgcn_s_wait_tensorcnt(0);
            }
        }
        __syncthreads();

        const _Float16* pa = lA[it & 1];
        const _Float16* pb = lB[it & 1];
        v16h af[4], bf[2];
#pragma unroll
        for (int mi = 0; mi < 4; ++mi) af[mi] = load_frag(&pa[(wm + mi * 16) * 40], 40);
#pragma unroll
        for (int ni = 0; ni < 2; ++ni) bf[ni] = load_frag(&pb[(wn + ni * 16) * 40], 40);
#pragma unroll
        for (int mi = 0; mi < 4; ++mi)
#pragma unroll
            for (int ni = 0; ni < 2; ++ni)
                acc[mi][ni] = wmma_f16(af[mi], bf[ni], acc[mi][ni]);
        __syncthreads();   // reads done before TDM may overwrite this buffer
    }

    const int cn = lane & 15, ch = lane >> 4;
#pragma unroll
    for (int mi = 0; mi < 4; ++mi)
#pragma unroll
        for (int ni = 0; ni < 2; ++ni)
#pragma unroll
            for (int r = 0; r < 8; ++r) {
                const int gm = bm + wm + mi * 16 + r + ch * 8;
                const int gn = bn + wn + ni * 16 + cn;
                const float v = acc[mi][ni][r];
                if (MODE == 0) {
                    const int which = gn >> 10;          // 0=q 1=k 2=v
                    const int ic    = gn & 1023;
                    const int head  = ic >> 6;
                    const int dh    = ic & 63;
                    const int bb    = gm >> 11;
                    const int seq   = gm & 2047;
                    const size_t hb = (size_t)bb * 16 + head;
                    if (which == 0)
                        Qb[(hb * 2048 + seq) * 64 + dh] = (_Float16)(v * 0.125f);
                    else if (which == 1)
                        Kb[(hb * 2048 + seq) * 64 + dh] = (_Float16)v;
                    else                                  // V stored transposed
                        Vt[(hb * 64 + dh) * 2048 + seq] = (_Float16)v;
                } else {
                    Out[(size_t)gm * 1024 + gn] = v;
                }
            }
}

// ---------------------------------------------------------------------------
// 4) Flash attention per (b,h): TDM-staged K/Vt tiles, online softmax with
//    DPP row reductions. Block = 8 waves x 16 rows; column tiles of 64.
// ---------------------------------------------------------------------------
__global__ __launch_bounds__(256)
void attn_flash(const _Float16* __restrict__ Qb, const _Float16* __restrict__ Kb,
                const _Float16* __restrict__ Vt,
                const unsigned char* __restrict__ mask,
                _Float16* __restrict__ AO) {
    __shared__ __align__(16) _Float16 lK[2][64 * 72];    // [j][d], 64+8 pad
    __shared__ __align__(16) _Float16 lV[2][64 * 72];    // [d][j], 64+8 pad
    __shared__ __align__(16) _Float16 lP[8][16 * 72];    // per-wave P [i][j]

    const int tid  = threadIdx.x;
    const int wave = tid >> 5;
    const int lane = tid & 31;
    const int cn   = lane & 15;
    const int ch   = lane >> 4;

    const int bh = blockIdx.x;                 // 0..31
    const int b  = bh >> 4;
    const int bi = blockIdx.y;                 // row block 0..15
    const int h  = bh & 15;
    const int rowBase = bi * 128 + wave * 16;
    const size_t kb64 = (size_t)bh * 2048 * 64;   // Q/K base (elements)
    const size_t vb64 = (size_t)bh * 64 * 2048;   // Vt base (elements)

    v16h qf0 = load_frag(Qb + kb64 + (size_t)rowBase * 64, 64);
    v16h qf1 = load_frag(Qb + kb64 + (size_t)rowBase * 64 + 32, 64);

    v8f o[4] = {};
    float m_r[8], l_r[8];
#pragma unroll
    for (int r = 0; r < 8; ++r) { m_r[r] = -3.0e38f; l_r[r] = 0.f; }

    const int ntiles = 2 * bi + 2;             // causal
    // prologue: tile 0 (row=64 halfs=128B=16x8B -> pi=4, 16B pad -> pa=3)
    if (wave == 0) {
        tdm_load_2d(lds_off_u32(&lK[0][0]), Kb + kb64, 64, 2048, 64, 64, 64, 4, 3);
        tdm_load_2d(lds_off_u32(&lV[0][0]), Vt + vb64, 2048, 64, 2048, 64, 64, 4, 3);
    }

    for (int jt = 0; jt < ntiles; ++jt) {
        const int jb = jt * 64;
        if (wave == 0) {
            if (jt + 1 < ntiles) {
                const int jn = jb + 64;
                tdm_load_2d(lds_off_u32(&lK[(jt + 1) & 1][0]),
                            Kb + kb64 + (size_t)jn * 64, 64, 2048, 64, 64, 64, 4, 3);
                tdm_load_2d(lds_off_u32(&lV[(jt + 1) & 1][0]),
                            Vt + vb64 + jn, 2048, 64, 2048, 64, 64, 4, 3);
                __builtin_amdgcn_s_wait_tensorcnt(2);
            } else {
                __builtin_amdgcn_s_wait_tensorcnt(0);
            }
        }
        __syncthreads();
        const _Float16* pk = lK[jt & 1];
        const _Float16* pv = lV[jt & 1];

        // S = Q @ K^T (Q pre-scaled)
        v8f s[4];
#pragma unroll
        for (int nf = 0; nf < 4; ++nf) {
            v8f t = {};
            v16h b0 = load_frag(&pk[(nf * 16) * 72], 72);
            v16h b1 = load_frag(&pk[(nf * 16) * 72 + 32], 72);
            t = wmma_f16(qf0, b0, t);
            t = wmma_f16(qf1, b1, t);
            s[nf] = t;
        }

        // causal + padding mask, per-row max
        float mt[8];
#pragma unroll
        for (int r = 0; r < 8; ++r) mt[r] = -3.0e38f;
#pragma unroll
        for (int nf = 0; nf < 4; ++nf)
#pragma unroll
            for (int r = 0; r < 8; ++r) {
                const int gi = rowBase + r + ch * 8;
                const int gj = jb + nf * 16 + cn;
                const bool ok = (gj <= gi) && (mask[b * 2048 + gj] != 0);
                const float sv = ok ? s[nf][r] : -3.0e38f;
                s[nf][r] = sv;
                mt[r] = fmaxf(mt[r], sv);
            }
#pragma unroll
        for (int r = 0; r < 8; ++r) mt[r] = xmax16(mt[r]);

        float alpha[8], rs[8];
#pragma unroll
        for (int r = 0; r < 8; ++r) {
            const float mn = fmaxf(m_r[r], mt[r]);
            alpha[r] = __expf(m_r[r] - mn);
            m_r[r]   = mn;
            rs[r]    = 0.f;
        }

        // P = exp(S - m) -> LDS (re-layout C-frag -> A-frag)
        _Float16* pW = &lP[wave][0];
#pragma unroll
        for (int nf = 0; nf < 4; ++nf)
#pragma unroll
            for (int r = 0; r < 8; ++r) {
                const float p = __expf(s[nf][r] - m_r[r]);
                rs[r] += p;
                pW[(r + ch * 8) * 72 + nf * 16 + cn] = (_Float16)p;
            }
#pragma unroll
        for (int r = 0; r < 8; ++r) {
            rs[r] = xsum16(rs[r]);
            l_r[r] = l_r[r] * alpha[r] + rs[r];
        }
#pragma unroll
        for (int df = 0; df < 4; ++df)
#pragma unroll
            for (int r = 0; r < 8; ++r) o[df][r] *= alpha[r];

        // same-wave LDS RAW on P (split dependency counters)
        asm volatile("s_wait_dscnt 0" ::: "memory");

        v16h pa0 = load_frag(pW, 72);
        v16h pa1 = load_frag(pW + 32, 72);
#pragma unroll
        for (int df = 0; df < 4; ++df) {
            v16h vb0 = load_frag(&pv[(df * 16) * 72], 72);
            v16h vb1 = load_frag(&pv[(df * 16) * 72 + 32], 72);
            o[df] = wmma_f16(pa0, vb0, o[df]);
            o[df] = wmma_f16(pa1, vb1, o[df]);
        }
        __syncthreads();   // tile reads done before TDM overwrite
    }

#pragma unroll
    for (int df = 0; df < 4; ++df)
#pragma unroll
        for (int r = 0; r < 8; ++r) {
            const int gi = rowBase + r + ch * 8;
            const float val = o[df][r] / l_r[r];
            AO[((size_t)b * 2048 + gi) * 1024 + h * 64 + df * 16 + cn] =
                (_Float16)val;
        }
}

// ---------------------------------------------------------------------------
// launch
// ---------------------------------------------------------------------------
extern "C" void kernel_launch(void* const* d_in, const int* in_sizes, int n_in,
                              void* d_out, int out_size, void* d_ws, size_t ws_size,
                              hipStream_t stream) {
    const float*         x     = (const float*)d_in[0];
    const unsigned char* mask  = (const unsigned char*)d_in[1];
    const float*         gamma = (const float*)d_in[2];
    const float*         beta  = (const float*)d_in[3];
    const float*         w_qkv = (const float*)d_in[4];
    const float*         w_out = (const float*)d_in[5];
    float*               out   = (float*)d_out;

    _Float16* ws    = (_Float16*)d_ws;
    _Float16* xnAO  = ws;                                  // 4096*1024 (xn, later AO)
    _Float16* wqkvT = xnAO  + (size_t)4096 * 1024;         // 3072*1024
    _Float16* woutT = wqkvT + (size_t)3072 * 1024;         // 1024*1024
    _Float16* Qb    = woutT + (size_t)1024 * 1024;         // 4096*1024
    _Float16* Kb    = Qb    + (size_t)4096 * 1024;
    _Float16* Vt    = Kb    + (size_t)4096 * 1024;         // ~40 MB total

    ln_kernel<<<4096, 256, 0, stream>>>(x, gamma, beta, xnAO);
    transpose_to_f16<<<(3072 * 1024) / 256, 256, 0, stream>>>(w_qkv, wqkvT, 3072);
    transpose_to_f16<<<(1024 * 1024) / 256, 256, 0, stream>>>(w_out, woutT, 1024);

    gemm_wmma<0><<<dim3(24, 32), 256, 0, stream>>>(xnAO, wqkvT, Qb, Kb, Vt, nullptr);
    attn_flash<<<dim3(32, 16), 256, 0, stream>>>(Qb, Kb, Vt, mask, xnAO);
    gemm_wmma<1><<<dim3(8, 32), 256, 0, stream>>>(xnAO, woutT, nullptr, nullptr,
                                                  nullptr, out);
}
